// MultiHeadSelfAttention_1614907703876
// MI455X (gfx1250) — compile-verified, matrix operand order checked by reading, not run on hardware
//
#include <hip/hip_runtime.h>
#include <hip/hip_bf16.h>

// ---------------------------------------------------------------------------
// MHA for MI455X (gfx1250): all GEMM-shaped ops via v_wmma_f32_16x16x32_f16
// (wave32), f16 data / f32 accumulate, software-pipelined fragment loads.
// ---------------------------------------------------------------------------

typedef __attribute__((ext_vector_type(16))) _Float16 v16h;
typedef __attribute__((ext_vector_type(8)))  _Float16 v8h;
typedef __attribute__((ext_vector_type(4)))  _Float16 v4h;
typedef __attribute__((ext_vector_type(8)))  float    v8f;
typedef __attribute__((ext_vector_type(4)))  float    v4f;

#define WMMA_F16(a, b, c) \
  __builtin_amdgcn_wmma_f32_16x16x32_f16(false, (a), false, (b), (short)0, (c), false, false)

__device__ __forceinline__ v16h combine8(v8h lo, v8h hi) {
  return __builtin_shufflevector(lo, hi, 0,1,2,3,4,5,6,7,8,9,10,11,12,13,14,15);
}

// A/B fragment loader for 16-bit WMMA (16x16x32). For lane half = lane/16,
// the lane's 16 f16 values are two contiguous 8-element K-chunks:
//   [k0 + half*8, +8)  and  [k0 + 16 + half*8, +8)
__device__ __forceinline__ v16h load_frag(const _Float16* __restrict__ rowptr,
                                          int k0, int half) {
  const _Float16* p = rowptr + k0 + half * 8;
  v8h lo = *(const v8h*)p;
  v8h hi = *(const v8h*)(p + 16);
  return combine8(lo, hi);
}

// ---------------------------------------------------------------------------
// fp32 -> f16 cast, 4 elements per thread (b128 load / b64 store)
// ---------------------------------------------------------------------------
__global__ void cast_f32_f16_kernel(const float* __restrict__ src,
                                    _Float16* __restrict__ dst, int n4) {
  int i = blockIdx.x * blockDim.x + threadIdx.x;
  if (i < n4) {
    v4f v = ((const v4f*)src)[i];
    v4h h;
    h.x = (_Float16)v.x; h.y = (_Float16)v.y;
    h.z = (_Float16)v.z; h.w = (_Float16)v.w;
    ((v4h*)dst)[i] = h;
  }
}

// ---------------------------------------------------------------------------
// y = A @ W^T   (A: [4096,1024] f16 row-major, W: [1024,1024] f16 row-major)
// One wave computes a 16(M) x 64(N) strip; fragment loads are double-buffered
// so each iteration's 10 b128 loads overlap the previous iteration's WMMAs.
// mode 0: f16 -> [B,H,T,Dh]   (Q / K layout)
// mode 1: f16 -> [B,H,Dh,T]   (V transposed, for PV B-fragments)
// mode 2: f32 -> row-major [4096,1024] (final output)
// ---------------------------------------------------------------------------
__global__ void __launch_bounds__(128)
gemm_xwT_kernel(const _Float16* __restrict__ A, const _Float16* __restrict__ W,
                void* __restrict__ out, int mode) {
  const int lane = threadIdx.x & 31;
  const int half = lane >> 4;
  const int ln   = lane & 15;
  const int wid  = blockIdx.x * 4 + (threadIdx.x >> 5);   // 4096 waves total
  const int mt   = wid >> 4;                              // 0..255 (M/16)
  const int nt   = wid & 15;                              // 0..15  (N/64)
  const int mbase = mt * 16;
  const int nbase = nt * 64;

  const _Float16* arow = A + (size_t)(mbase + ln) * 1024;
  const _Float16* w0   = W + (size_t)(nbase + ln) * 1024;
  const _Float16* w1   = w0 + 16 * 1024;
  const _Float16* w2   = w1 + 16 * 1024;
  const _Float16* w3   = w2 + 16 * 1024;

  v8f acc0 = {}, acc1 = {}, acc2 = {}, acc3 = {};

  // prologue: fragments for k0 = 0
  v16h af  = load_frag(arow, 0, half);
  v16h bf0 = load_frag(w0, 0, half);
  v16h bf1 = load_frag(w1, 0, half);
  v16h bf2 = load_frag(w2, 0, half);
  v16h bf3 = load_frag(w3, 0, half);

  for (int k0 = 32; k0 < 1024; k0 += 32) {
    // issue next k-step's loads first so they overlap the WMMAs below
    v16h an  = load_frag(arow, k0, half);
    v16h bn0 = load_frag(w0, k0, half);
    v16h bn1 = load_frag(w1, k0, half);
    v16h bn2 = load_frag(w2, k0, half);
    v16h bn3 = load_frag(w3, k0, half);

    acc0 = WMMA_F16(af, bf0, acc0);
    acc1 = WMMA_F16(af, bf1, acc1);
    acc2 = WMMA_F16(af, bf2, acc2);
    acc3 = WMMA_F16(af, bf3, acc3);

    af = an; bf0 = bn0; bf1 = bn1; bf2 = bn2; bf3 = bn3;
  }
  // epilogue k-step
  acc0 = WMMA_F16(af, bf0, acc0);
  acc1 = WMMA_F16(af, bf1, acc1);
  acc2 = WMMA_F16(af, bf2, acc2);
  acc3 = WMMA_F16(af, bf3, acc3);

  v8f acc[4] = {acc0, acc1, acc2, acc3};
#pragma unroll
  for (int j = 0; j < 4; ++j) {
#pragma unroll
    for (int r = 0; r < 8; ++r) {
      float v = acc[j][r];
      int gm = mbase + r + 8 * half;       // token row (b*2048 + t)
      int gn = nbase + j * 16 + ln;        // feature (h*64 + d)
      if (mode == 0) {
        int b = gm >> 11, t = gm & 2047, h = gn >> 6, d = gn & 63;
        ((_Float16*)out)[(((size_t)b * 16 + h) * 2048 + t) * 64 + d] = (_Float16)v;
      } else if (mode == 1) {
        int b = gm >> 11, t = gm & 2047, h = gn >> 6, d = gn & 63;
        ((_Float16*)out)[(((size_t)b * 16 + h) * 64 + d) * 2048 + t] = (_Float16)v;
      } else {
        ((float*)out)[(size_t)gm * 1024 + gn] = v;
      }
    }
  }
}

// ---------------------------------------------------------------------------
// Flash-style causal attention. One wave per (b,h, 16-query tile).
// Q,K: [B,H,T,64] f16;  Vt: [B,H,64,T] f16;  Yh out: [B,T,1024] f16.
// K fragments are double-buffered across key blocks; V fragments are issued
// before the softmax VALU work so they overlap it.
// ---------------------------------------------------------------------------
__global__ void __launch_bounds__(32)
attn_kernel(const _Float16* __restrict__ Q, const _Float16* __restrict__ K,
            const _Float16* __restrict__ Vt, _Float16* __restrict__ Yh) {
  const int lane = threadIdx.x;
  const int half = lane >> 4;
  const int ln   = lane & 15;
  const int qt   = blockIdx.x & 127;   // query tile (T/16 = 128)
  const int bh   = blockIdx.x >> 7;    // b*16 + h
  const int hh   = bh & 15;
  const int bb   = bh >> 4;

  const _Float16* Qbh = Q  + (size_t)bh * 2048 * 64;
  const _Float16* Kbh = K  + (size_t)bh * 2048 * 64;
  const _Float16* Vbh = Vt + (size_t)bh * 64 * 2048;

  // Preload Q fragments: A rows = 16 query rows, K-dim = 64 (two 32-chunks)
  const _Float16* qrow = Qbh + (size_t)(qt * 16 + ln) * 64;
  v16h qf0 = load_frag(qrow, 0, half);
  v16h qf1 = load_frag(qrow, 32, half);

  float mrow[8], lrow[8];
#pragma unroll
  for (int r = 0; r < 8; ++r) { mrow[r] = -1e30f; lrow[r] = 0.0f; }
  v8f acc[4] = {};

  __shared__ __attribute__((aligned(16))) _Float16 plds[16 * 32];

  const int nkb = qt / 2 + 1;   // number of 32-key blocks (causal)

  // prologue: K fragments for key block 0
  v16h kf00, kf01, kf10, kf11;
  {
    const _Float16* krow0 = Kbh + (size_t)ln * 64;
    const _Float16* krow1 = Kbh + (size_t)(16 + ln) * 64;
    kf00 = load_frag(krow0, 0, half);
    kf01 = load_frag(krow0, 32, half);
    kf10 = load_frag(krow1, 0, half);
    kf11 = load_frag(krow1, 32, half);
  }

  for (int jb = 0; jb < nkb; ++jb) {
    const int kb0 = jb * 32;

    // V fragments: independent of S/softmax -> issue first, consume last
    v16h vf0 = load_frag(Vbh + (size_t)(0  + ln) * 2048, kb0, half);
    v16h vf1 = load_frag(Vbh + (size_t)(16 + ln) * 2048, kb0, half);
    v16h vf2 = load_frag(Vbh + (size_t)(32 + ln) * 2048, kb0, half);
    v16h vf3 = load_frag(Vbh + (size_t)(48 + ln) * 2048, kb0, half);

    // S = Q @ K^T for 32 keys: two 16x16 tiles, each over K-dim 64
    v8f s0 = {}, s1 = {};
    s0 = WMMA_F16(qf0, kf00, s0);
    s0 = WMMA_F16(qf1, kf01, s0);
    s1 = WMMA_F16(qf0, kf10, s1);
    s1 = WMMA_F16(qf1, kf11, s1);

    // prefetch next block's K fragments (clamped on the last iteration)
    {
      const int kbn = (jb + 1 < nkb) ? kb0 + 32 : kb0;
      const _Float16* krow0 = Kbh + (size_t)(kbn + ln) * 64;
      const _Float16* krow1 = Kbh + (size_t)(kbn + 16 + ln) * 64;
      kf00 = load_frag(krow0, 0, half);
      kf01 = load_frag(krow0, 32, half);
      kf10 = load_frag(krow1, 0, half);
      kf11 = load_frag(krow1, 32, half);
    }

    // Scale, causal mask, online softmax (per C-layout row r+8*half)
#pragma unroll
    for (int r = 0; r < 8; ++r) {
      const int qg = qt * 16 + r + 8 * half;
      float v0 = s0[r] * 0.125f;
      float v1 = s1[r] * 0.125f;
      if (kb0 + ln      > qg) v0 = -1e30f;
      if (kb0 + 16 + ln > qg) v1 = -1e30f;

      float mx = fmaxf(v0, v1);
#pragma unroll
      for (int off = 1; off < 16; off <<= 1)
        mx = fmaxf(mx, __shfl_xor(mx, off, 32));
      const float mn = fmaxf(mrow[r], mx);
      const float al = __expf(mrow[r] - mn);
      const float p0 = __expf(v0 - mn);
      const float p1 = __expf(v1 - mn);
      float sm = p0 + p1;
#pragma unroll
      for (int off = 1; off < 16; off <<= 1)
        sm += __shfl_xor(sm, off, 32);
      lrow[r] = lrow[r] * al + sm;
      mrow[r] = mn;
#pragma unroll
      for (int j = 0; j < 4; ++j) acc[j][r] *= al;

      // stash P in row-major 16x32 f16 for the A-fragment re-layout
      plds[(r + 8 * half) * 32 + ln]      = (_Float16)p0;
      plds[(r + 8 * half) * 32 + 16 + ln] = (_Float16)p1;
    }

    // LDS in-order per wave; fence compiler + wait DS counter before re-read
    asm volatile("s_wait_dscnt 0" ::: "memory");

    // P as A-fragment: row m = ln, K-chunks at half*8 and 16+half*8
    v16h pf = combine8(*(const v8h*)&plds[ln * 32 + half * 8],
                       *(const v8h*)&plds[ln * 32 + 16 + half * 8]);

    // Y += P @ V
    acc[0] = WMMA_F16(pf, vf0, acc[0]);
    acc[1] = WMMA_F16(pf, vf1, acc[1]);
    acc[2] = WMMA_F16(pf, vf2, acc[2]);
    acc[3] = WMMA_F16(pf, vf3, acc[3]);
  }

  // Normalize and store into [B,T,1024] f16 for the output projection
#pragma unroll
  for (int j = 0; j < 4; ++j) {
#pragma unroll
    for (int r = 0; r < 8; ++r) {
      float v = acc[j][r] / lrow[r];
      int t = qt * 16 + r + 8 * half;
      int c = hh * 64 + j * 16 + ln;
      Yh[((size_t)bb * 2048 + t) * 1024 + c] = (_Float16)v;
    }
  }
}

// ---------------------------------------------------------------------------
extern "C" void kernel_launch(void* const* d_in, const int* in_sizes, int n_in,
                              void* d_out, int out_size, void* d_ws, size_t ws_size,
                              hipStream_t stream) {
  const float* x  = (const float*)d_in[0];
  const float* Wq = (const float*)d_in[1];
  const float* Wk = (const float*)d_in[2];
  const float* Wv = (const float*)d_in[3];
  const float* Wo = (const float*)d_in[4];

  const int MT = 4096;          // B*T
  const int C  = 1024;

  _Float16* ws  = (_Float16*)d_ws;
  _Float16* xh  = ws;                    // 4M
  _Float16* wqh = xh  + (size_t)MT * C;  // 1M each
  _Float16* wkh = wqh + (size_t)C * C;
  _Float16* wvh = wkh + (size_t)C * C;
  _Float16* woh = wvh + (size_t)C * C;
  _Float16* Qb  = woh + (size_t)C * C;   // [B,H,T,64]
  _Float16* Kb  = Qb  + (size_t)MT * C;  // [B,H,T,64]
  _Float16* Vtb = Kb  + (size_t)MT * C;  // [B,H,64,T]
  _Float16* yh  = Vtb + (size_t)MT * C;  // [B,T,1024]

  // Casts (vectorized x4)
  cast_f32_f16_kernel<<<(MT * C / 4 + 255) / 256, 256, 0, stream>>>(x, xh, MT * C / 4);
  cast_f32_f16_kernel<<<(C * C / 4 + 255) / 256, 256, 0, stream>>>(Wq, wqh, C * C / 4);
  cast_f32_f16_kernel<<<(C * C / 4 + 255) / 256, 256, 0, stream>>>(Wk, wkh, C * C / 4);
  cast_f32_f16_kernel<<<(C * C / 4 + 255) / 256, 256, 0, stream>>>(Wv, wvh, C * C / 4);
  cast_f32_f16_kernel<<<(C * C / 4 + 255) / 256, 256, 0, stream>>>(Wo, woh, C * C / 4);

  // Projections: 4096 waves = 1024 blocks x 4 waves each
  gemm_xwT_kernel<<<1024, 128, 0, stream>>>(xh, wqh, (void*)Qb, 0);
  gemm_xwT_kernel<<<1024, 128, 0, stream>>>(xh, wkh, (void*)Kb, 0);
  gemm_xwT_kernel<<<1024, 128, 0, stream>>>(xh, wvh, (void*)Vtb, 1);

  // Attention: one wave per (b,h,query-tile) -> 2*16*128 = 4096 blocks
  attn_kernel<<<4096, 32, 0, stream>>>(Qb, Kb, Vtb, yh);

  // Output projection straight to fp32 d_out
  gemm_xwT_kernel<<<1024, 128, 0, stream>>>(yh, woh, d_out, 2);
}
